// BasicRecurrentEntityEncoder_8770323219173
// MI455X (gfx1250) — matrix operand order repaired and checked by reading, not compiled, not run on hardware
//
#include <hip/hip_runtime.h>

// CDNA5 / gfx1250. One workgroup per batch element; recurrent h@U done with
// v_wmma_f32_16x16x32_bf16 on LDS-resident state. keys@V precomputed once.

typedef __attribute__((ext_vector_type(16))) __bf16 v16bf;
typedef __attribute__((ext_vector_type(8)))  float  v8f;

#define BB 256
#define SS 256
#define KK 64
#define DD 128

__global__ __launch_bounds__(256)
void entnet_kernel(const float* __restrict__ es,          // [B,S,D]
                   const unsigned char* __restrict__ msk, // [B,S] bool
                   const float* __restrict__ keys,        // [B,K,D]
                   const float* __restrict__ U,           // [D,D]
                   const float* __restrict__ V,           // [D,D]
                   const float* __restrict__ W,           // [D,D]
                   float* __restrict__ out)               // [B,K,D]
{
    __shared__ float h_lds[KK * DD];                      // 32KB master h (f32)
    __shared__ float kv_lds[KK * DD];                     // 32KB keys@V (f32)
    __shared__ __align__(32) unsigned char uw_buf[DD * DD * 4]; // 64KB: V(f32) staging, then Ut|Wt (bf16)
    __shared__ float s_lds[DD];
    __shared__ float g_lds[KK];
    __shared__ float sw_lds[DD];

    __bf16* ut = (__bf16*)uw_buf;                 // Ut[n][k] = U[k][n], bf16, 32KB
    __bf16* wt = (__bf16*)(uw_buf + DD * DD * 2); // Wt[n][d] = W[d][n], bf16, 32KB
    float*  vtmp = (float*)uw_buf;                // V staging, f32, 64KB

    const int b    = blockIdx.x;
    const int tid  = threadIdx.x;
    const int lane = tid & 31;
    const int wave = tid >> 5;
    const float* keysb = keys + (size_t)b * KK * DD;

    // ---------------- init ----------------
    // Stage keys[b] into h_lds (temporarily) and V into vtmp.
    for (int i = tid; i < KK * DD; i += 256) h_lds[i] = keysb[i];
    for (int i = tid; i < DD * DD; i += 256) vtmp[i]  = V[i];
    __syncthreads();

    // kv = keys @ V  (once; time-invariant)
    {
        const int k  = tid >> 2;
        const int n0 = (tid & 3) * 32;
        float acc[32];
#pragma unroll
        for (int j = 0; j < 32; ++j) acc[j] = 0.f;
        for (int d = 0; d < DD; ++d) {
            const float kd = h_lds[k * DD + d];
            const float* vr = &vtmp[d * DD + n0];
#pragma unroll
            for (int j = 0; j < 32; ++j) acc[j] += kd * vr[j];
        }
#pragma unroll
        for (int j = 0; j < 32; ++j) kv_lds[k * DD + n0 + j] = acc[j];
    }
    __syncthreads();

    // Build Ut, Wt (bf16, transposed) over the V staging area; zero h.
    for (int i = tid; i < DD * DD; i += 256) {
        const int r = i >> 7, c = i & 127;
        ut[c * DD + r] = (__bf16)U[i];
    }
    for (int i = tid; i < DD * DD; i += 256) {
        const int r = i >> 7, c = i & 127;
        wt[c * DD + r] = (__bf16)W[i];
    }
    for (int i = tid; i < KK * DD; i += 256) h_lds[i] = 0.f;
    __syncthreads();

    // ---------------- recurrence ----------------
    const int mt = wave; // M-tile for WMMA waves 0..3
    for (int t = 0; t < SS; ++t) {
        if (tid < DD) s_lds[tid] = es[((size_t)b * SS + t) * DD + tid];
        __syncthreads(); // h stable, s ready

        v8f acc[8] = {};
        if (wave < 4) {
            // ---- h @ U via WMMA: rows mt*16 .. mt*16+15, all 8 N-tiles ----
            const int hi   = lane >> 4;
            const int lrow = lane & 15;
            const int row  = mt * 16 + lrow; // A: both lane halves carry M = lane%16
#pragma unroll
            for (int kt = 0; kt < 4; ++kt) {
                // A fragment (16x32 bf16): lane half selects K offsets {0..15} or {16..31}
                v16bf a;
                const float* hr = &h_lds[row * DD + kt * 32];
                const int o1 = hi ? 16 : 0;
#pragma unroll
                for (int i = 0; i < 8; ++i) {
                    a[i]     = (__bf16)hr[o1 + i];
                    a[8 + i] = (__bf16)hr[8 + o1 + i];
                }
                const int kb = kt * 32 + (hi ? 16 : 0);
#pragma unroll
                for (int nt = 0; nt < 8; ++nt) {
                    const int n = nt * 16 + lrow;
                    v16bf bb = *(const v16bf*)&ut[n * DD + kb]; // K-contiguous 32B
                    acc[nt] = __builtin_amdgcn_wmma_f32_16x16x32_bf16(
                        false, a, false, bb, (short)0, acc[nt], false, false);
                }
            }
        } else {
            // ---- gate g[k] = sigmoid(s . (h_k + keys_k)) ; sw = s @ W ----
            const int tid2 = tid - 128;          // 0..127
            const int k    = tid2 >> 1;
            const int half = tid2 & 1;
            const float* kr = keysb + k * DD + half * 64;
            const float* hr = &h_lds[k * DD + half * 64];
            const float* sr = &s_lds[half * 64];
            float a = 0.f;
#pragma unroll 16
            for (int i = 0; i < 64; ++i) a += sr[i] * (hr[i] + kr[i]);
            a += __shfl_xor(a, 1, 32);
            if (!half) g_lds[k] = 1.f / (1.f + __expf(-a));

            const int n = tid2;
            float sacc = 0.f;
#pragma unroll 16
            for (int d = 0; d < DD; ++d) sacc += s_lds[d] * (float)wt[n * DD + d];
            sw_lds[n] = sacc;
        }
        __syncthreads(); // acc done, g/sw ready, all h reads done

        if (wave < 4) {
            // ---- epilogue on accumulator registers ----
            const unsigned char mb = msk[b * SS + t];
            const int hi = lane >> 4;
            const int c  = lane & 15;
#pragma unroll
            for (int r = 0; r < 8; ++r) {
                const int row = mt * 16 + r + hi * 8; // C/D layout: VGPR r -> M=r (lo) / r+8 (hi)
                const float gv = g_lds[row];
                float upd[8], hold[8];
                float ssq = 0.f;
#pragma unroll
                for (int nt = 0; nt < 8; ++nt) {
                    const int col = nt * 16 + c;
                    const float pre = acc[nt][r] + kv_lds[row * DD + col] + sw_lds[col];
                    const float ht  = fmaxf(pre, 0.f); // relu
                    hold[nt] = h_lds[row * DD + col];
                    upd[nt]  = hold[nt] + gv * ht;
                    ssq += upd[nt] * upd[nt];
                }
                // row sum-of-squares across the 16 lanes of this half-wave
                ssq += __shfl_xor(ssq, 1, 32);
                ssq += __shfl_xor(ssq, 2, 32);
                ssq += __shfl_xor(ssq, 4, 32);
                ssq += __shfl_xor(ssq, 8, 32);
                const float rn = __frsqrt_rn(fmaxf(ssq, 1e-12f));
#pragma unroll
                for (int nt = 0; nt < 8; ++nt) {
                    const int col = nt * 16 + c;
                    h_lds[row * DD + col] = mb ? upd[nt] * rn : hold[nt];
                }
            }
        }
        // loop-top barrier makes h writes visible before next step's reads
    }

    __syncthreads();
    float* ob = out + (size_t)b * KK * DD;
    for (int i = tid; i < KK * DD; i += 256) ob[i] = h_lds[i];
}

extern "C" void kernel_launch(void* const* d_in, const int* in_sizes, int n_in,
                              void* d_out, int out_size, void* d_ws, size_t ws_size,
                              hipStream_t stream) {
    (void)in_sizes; (void)n_in; (void)d_ws; (void)ws_size; (void)out_size;
    const float*         es   = (const float*)d_in[0];
    const unsigned char* mask = (const unsigned char*)d_in[1]; // jax bool -> 1 byte
    const float*         keys = (const float*)d_in[2];
    const float*         U    = (const float*)d_in[3];
    const float*         V    = (const float*)d_in[4];
    const float*         W    = (const float*)d_in[5];
    float*               out  = (float*)d_out;

    entnet_kernel<<<dim3(BB), dim3(256), 0, stream>>>(es, mask, keys, U, V, W, out);
}
